// NetSparsePacked_72980084293821
// MI455X (gfx1250) — compile-verified, matrix operand order checked by reading, not run on hardware
//
#include <hip/hip_runtime.h>

typedef __attribute__((ext_vector_type(16))) _Float16 v16h;
typedef __attribute__((ext_vector_type(8)))  float    v8f;

union Frag {
  v16h v;
  uint4 q[2];
};

#define BATCH  32768
#define K1     784      // = 98*8
#define N1     256
#define MBLK   64
#define KTILE  32
#define NKT    25       // K padded to 800
#define KS     40       // padded LDS row stride (halves) for xT/wT: 80B -> conflict-free b128
#define HS     264      // padded LDS row stride (halves) for hT/w2T: 528B -> conflict-free b128

#define W1T_ELEMS (NKT * N1 * KTILE)

// ---------------- pre-decode kernel (1 block, runs once, negligible) ----------------
__global__ __launch_bounds__(256)
void decode_weights(const int* __restrict__ w1p, const int* __restrict__ m1p,
                    const float* __restrict__ a1p,
                    const int* __restrict__ w2p, const int* __restrict__ m2p,
                    const float* __restrict__ a2p,
                    _Float16* __restrict__ w1t, _Float16* __restrict__ w2t)
{
  const int tid = threadIdx.x;
  const float alpha1 = *a1p;
  const float alpha2 = *a2p;

  // w1: thread tid owns output row o=tid; tile-major [25][256][32], K padded to 800
  const int o = tid;
  for (int tk = 0; tk < NKT; ++tk) {
#pragma unroll
    for (int g = 0; g < 4; ++g) {
      int byteIdx = tk * 4 + g;
      int wb = 0, mb = 0;
      if (byteIdx < 98) { wb = w1p[o * 98 + byteIdx]; mb = m1p[o * 98 + byteIdx]; }
#pragma unroll
      for (int j = 0; j < 8; ++j) {
        int bs = 7 - j;                                   // MSB-first
        float wv = ((wb >> bs) & 1) ? 1.f : -1.f;
        float mv = (float)((mb >> bs) & 1);
        w1t[(tk * N1 + o) * KTILE + g * 8 + j] = (_Float16)(wv * mv * alpha1);
      }
    }
  }

  // w2: [16][256] dense in ws, rows 10..15 zero-padded
  for (int idx = tid; idx < 6 * N1; idx += 256)
    w2t[10 * N1 + idx] = (_Float16)0.f;
  for (int it = tid; it < 10 * 32; it += 256) {
    int oo = it >> 5, byteIdx = it & 31;
    int wb = w2p[oo * 32 + byteIdx];
    int mb = m2p[oo * 32 + byteIdx];
#pragma unroll
    for (int j = 0; j < 8; ++j) {
      int bs = 7 - j;
      float wv = ((wb >> bs) & 1) ? 1.f : -1.f;
      float mv = (float)((mb >> bs) & 1);
      w2t[oo * N1 + byteIdx * 8 + j] = (_Float16)(wv * mv * alpha2);
    }
  }
}

// ---------------- fused main kernel ----------------
__global__ __launch_bounds__(256)
void mlp_main(const float* __restrict__ x,
              const _Float16* __restrict__ w1t,
              const _Float16* __restrict__ w2t,
              float* __restrict__ out)
{
  __shared__ __align__(16) _Float16 xT[MBLK][KS];    //  5.0 KB (padded)
  __shared__ __align__(16) _Float16 wT[N1][KS];      // 20.0 KB (padded)
  __shared__ __align__(16) _Float16 hT[MBLK][HS];    // 33.0 KB (padded)
  __shared__ __align__(16) _Float16 w2T[16][HS];     //  8.3 KB (padded)

  const int tid     = threadIdx.x;
  const int lane    = tid & 31;
  const int wave    = tid >> 5;
  const int rowBase = blockIdx.x * MBLK;

  // wave tiling: 4 waves in M (16 rows) x 2 waves in N (128 cols = 8 WMMA tiles)
  const int mSub = wave >> 1;
  const int nSub = wave & 1;
  const int sel  = lane >> 4;        // K half-group select (WMMA A/B layout)
  const int lan  = lane & 15;

  // ---- copy pre-decoded w2 into padded LDS rows: 16 rows x 256 halves ----
  {
    int row = tid >> 4;              // 16 threads per row
    int col = (tid & 15) * 16;       // 16 halves = 32B per thread
    const uint4* src = (const uint4*)(w2t + row * N1 + col);
    uint4* dst = (uint4*)&w2T[row][col];
    dst[0] = src[0];
    dst[1] = src[1];
  }

  // staging registers (software pipeline)
  const int xRow = tid >> 2;             // 64 rows, 4 threads per row
  const int xCol = (tid & 3) * 8;        // 8 floats per thread
  const float* xSrc = x + (size_t)(rowBase + xRow) * K1;

  float4 xr0, xr1;
  uint4  wr0, wr1, wr2, wr3;

  auto loadW = [&](int tk) {
    const uint4* ws = (const uint4*)w1t + (size_t)tk * (N1 * KTILE / 8) + tid * 4;
    wr0 = ws[0]; wr1 = ws[1]; wr2 = ws[2]; wr3 = ws[3];
  };
  auto loadX_full = [&](int tk) {        // tiles 0..23: always in-bounds
    int k = tk * KTILE + xCol;
    xr0 = *(const float4*)(xSrc + k);
    xr1 = *(const float4*)(xSrc + k + 4);
  };
  auto loadX_tail = [&]() {              // tile 24: k = 768+xCol, valid iff xCol < 16
    xr0 = make_float4(0.f, 0.f, 0.f, 0.f);
    xr1 = xr0;
    if (xCol < 16) {
      int k = 24 * KTILE + xCol;
      xr0 = *(const float4*)(xSrc + k);
      xr1 = *(const float4*)(xSrc + k + 4);
    }
  };
  auto commitTile = [&]() {
    _Float16* xd = &xT[xRow][xCol];
    xd[0] = (_Float16)xr0.x; xd[1] = (_Float16)xr0.y;
    xd[2] = (_Float16)xr0.z; xd[3] = (_Float16)xr0.w;
    xd[4] = (_Float16)xr1.x; xd[5] = (_Float16)xr1.y;
    xd[6] = (_Float16)xr1.z; xd[7] = (_Float16)xr1.w;
    uint4* wd = (uint4*)&wT[tid][0];
    wd[0] = wr0; wd[1] = wr1; wd[2] = wr2; wd[3] = wr3;
  };

  v8f acc[8] = {};

  loadX_full(0);
  loadW(0);
  for (int tk = 0; tk < NKT; ++tk) {
    __syncthreads();                     // previous compute done reading LDS
    commitTile();
    __syncthreads();
    int nt = tk + 1;
    if (nt < NKT) {                      // issue next HBM loads before compute
      loadW(nt);
      if (nt < NKT - 1) loadX_full(nt);
      else              loadX_tail();
    }

    Frag A;
    A.q[0] = *(const uint4*)&xT[mSub * 16 + lan][sel * 8];
    A.q[1] = *(const uint4*)&xT[mSub * 16 + lan][16 + sel * 8];

    // ping-pong B fragments: WMMA(t) overlaps ds_loads for t+1
    Frag Bf[2];
    {
      int n = nSub * 128 + lan;
      Bf[0].q[0] = *(const uint4*)&wT[n][sel * 8];
      Bf[0].q[1] = *(const uint4*)&wT[n][16 + sel * 8];
    }
#pragma unroll
    for (int t = 0; t < 8; ++t) {
      if (t + 1 < 8) {
        int n = nSub * 128 + (t + 1) * 16 + lan;
        Bf[(t + 1) & 1].q[0] = *(const uint4*)&wT[n][sel * 8];
        Bf[(t + 1) & 1].q[1] = *(const uint4*)&wT[n][16 + sel * 8];
      }
      acc[t] = __builtin_amdgcn_wmma_f32_16x16x32_f16(
          false, A.v, false, Bf[t & 1].v, (short)0, acc[t], false, false);
    }
  }

  // ---- relu + write h tile to LDS (D layout: vgpr r, lane l -> m=r+8*(l>>4), n=l&15) ----
  __syncthreads();
#pragma unroll
  for (int t = 0; t < 8; ++t) {
#pragma unroll
    for (int r = 0; r < 8; ++r) {
      int m = r + 8 * sel;
      int n = nSub * 128 + t * 16 + lan;
      float v = acc[t][r];
      hT[mSub * 16 + m][n] = (_Float16)(v > 0.f ? v : 0.f);
    }
  }
  __syncthreads();

  // ---- layer 2: waves with nSub==0 (4 waves) each own 16 rows; K=256 from LDS ----
  if (nSub == 0) {
    v8f acc2 = {};
#pragma unroll
    for (int k0 = 0; k0 < 256; k0 += 32) {
      Frag A;
      A.q[0] = *(const uint4*)&hT[mSub * 16 + lan][k0 + sel * 8];
      A.q[1] = *(const uint4*)&hT[mSub * 16 + lan][k0 + 16 + sel * 8];
      Frag B;
      B.q[0] = *(const uint4*)&w2T[lan][k0 + sel * 8];
      B.q[1] = *(const uint4*)&w2T[lan][k0 + 16 + sel * 8];
      acc2 = __builtin_amdgcn_wmma_f32_16x16x32_f16(
          false, A.v, false, B.v, (short)0, acc2, false, false);
    }
    if (lan < 10) {
#pragma unroll
      for (int r = 0; r < 8; ++r) {
        int m = r + 8 * sel;
        out[(size_t)(rowBase + mSub * 16 + m) * 10 + lan] = acc2[r];
      }
    }
  }
}

extern "C" void kernel_launch(void* const* d_in, const int* in_sizes, int n_in,
                              void* d_out, int out_size, void* d_ws, size_t ws_size,
                              hipStream_t stream) {
  const float* x   = (const float*)d_in[0];
  const int*   w1p = (const int*)  d_in[1];
  const int*   m1p = (const int*)  d_in[2];
  const float* a1  = (const float*)d_in[3];
  const int*   w2p = (const int*)  d_in[4];
  const int*   m2p = (const int*)  d_in[5];
  const float* a2  = (const float*)d_in[6];
  float* out = (float*)d_out;

  _Float16* w1t = (_Float16*)d_ws;
  _Float16* w2t = w1t + W1T_ELEMS;

  decode_weights<<<dim3(1), dim3(256), 0, stream>>>(w1p, m1p, a1, w2p, m2p, a2, w1t, w2t);
  mlp_main<<<dim3(BATCH / MBLK), dim3(256), 0, stream>>>(x, w1t, w2t, out);
}
